// MGMoEModel_70300024700990
// MI455X (gfx1250) — matrix operand order repaired and checked by reading, not compiled, not run on hardware
//
#include <hip/hip_runtime.h>
#include <cstdint>

// ---------------------------------------------------------------------------
// Problem constants (match reference)
// ---------------------------------------------------------------------------
static constexpr int Bsz = 8192;   // batch
static constexpr int INF = 8000;   // input features
static constexpr int Cc  = 16;     // chunks
static constexpr int Ss  = 500;    // chunk size (K, padded to 512)
static constexpr int Sp  = 512;    // padded chunk K
static constexpr int Oo  = 64;     // per-chunk out
static constexpr int Ff  = 1024;   // C*O
static constexpr int Ee  = 8;      // experts
static constexpr int Dd  = 512;    // expert/task dim
static constexpr int Tt  = 2;      // tasks
static constexpr float EPS = 1e-5f;

typedef __attribute__((ext_vector_type(16))) __bf16 v16bf;
typedef __attribute__((ext_vector_type(8)))  float  v8f;

union Frag32B { uint4 q[2]; v16bf v; };

__device__ __forceinline__ uint16_t f2bf(float f) {
    uint32_t u = __float_as_uint(f);
    uint32_t r = u + 0x7FFFu + ((u >> 16) & 1u);   // round-to-nearest-even
    return (uint16_t)(r >> 16);
}
// Pack two f32 -> two bf16 in one dword: RNE round in 32-bit regs, then one
// v_perm_b32 grabs the two high halves ([ua.b2,ua.b3,ub.b2,ub.b3]).
__device__ __forceinline__ uint32_t pk2bf(float a, float b) {
    uint32_t ua = __float_as_uint(a), ub = __float_as_uint(b);
    ua += 0x7FFFu + ((ua >> 16) & 1u);
    ub += 0x7FFFu + ((ub >> 16) & 1u);
    return __builtin_amdgcn_perm(ub, ua, 0x07060302u);
}
__device__ __forceinline__ float bf2f(uint16_t h) {
    return __uint_as_float(((uint32_t)h) << 16);
}
__device__ __forceinline__ float swishf(float v) {
    return v / (1.0f + __expf(-v));
}

// ---------------------------------------------------------------------------
// Batched bf16 WMMA GEMM:  C[z] = A[z] * Bt[z]^T (+bias), BN stats atomics.
// A row-major [M][K] (f32 converted on the fly when AF32, else bf16).
// Bt PRE-TRANSPOSED bf16 [N][K] so all LDS staging is b128.
// Block tile: 128(M) x NTILE(N) x 32(K); 256 threads = 8 waves (4x2),
// wave tile 32 x (NTILE/2): NFR = NTILE/32 n-frags, 2*NFR WMMAs per K-step.
// ---------------------------------------------------------------------------
template <bool AF32, int NTILE>
__global__ __launch_bounds__(256)
void gemm_bf16_wmma(const void* __restrict__ Aq, int lda, long long aBatch,
                    const uint16_t* __restrict__ Bt, int ldbK, long long bBatch,
                    float* __restrict__ Cq, int ldc, long long cBatch,
                    const float* __restrict__ bias, int biasBatch,
                    float* __restrict__ ssum, float* __restrict__ ssumsq, int statBatch,
                    int Kpad, int Klog)
{
    constexpr int NFR = NTILE / 32;                    // n-frags per wave
    __shared__ __align__(16) uint16_t lA[128 * 40];    // A tile row-major, stride 40
    __shared__ __align__(16) uint16_t lBt[NTILE * 40]; // B tile [n][k], stride 40

    const int tid  = threadIdx.x;
    const int z    = blockIdx.z;
    const long long tileM = (long long)blockIdx.x * 128;
    const int colTile = blockIdx.y * NTILE;

    const float*    Af = (const float*)Aq;
    const uint16_t* Ah = (const uint16_t*)Aq;
    const long long aOff = (long long)z * aBatch;
    const uint16_t* Bp = Bt + (long long)z * bBatch + (long long)colTile * ldbK;

    const int wave = tid >> 5, lane = tid & 31;
    const int wm = wave >> 1, wn = wave & 1;

    v8f acc[2][NFR] = {};

    const int ar = tid >> 1;          // staged A row 0..127
    const int ak = (tid & 1) * 16;    // staged A k-offset (0|16)
    const long long aRowBase = (tileM + ar) * (long long)lda + aOff;

    for (int kb = 0; kb < Kpad; kb += 32) {
        // ---- stage A ----
        if constexpr (AF32) {
            const float* arow = Af + aRowBase;
            if (kb + 32 <= Klog) {              // uniform fast path
                const float4 f0 = *(const float4*)(arow + kb + ak);
                const float4 f1 = *(const float4*)(arow + kb + ak + 4);
                const float4 f2 = *(const float4*)(arow + kb + ak + 8);
                const float4 f3 = *(const float4*)(arow + kb + ak + 12);
                uint4 o0, o1;
                o0.x = pk2bf(f0.x, f0.y); o0.y = pk2bf(f0.z, f0.w);
                o0.z = pk2bf(f1.x, f1.y); o0.w = pk2bf(f1.z, f1.w);
                o1.x = pk2bf(f2.x, f2.y); o1.y = pk2bf(f2.z, f2.w);
                o1.z = pk2bf(f3.x, f3.y); o1.w = pk2bf(f3.z, f3.w);
                *(uint4*)(lA + ar * 40 + ak)     = o0;
                *(uint4*)(lA + ar * 40 + ak + 8) = o1;
            } else {                             // rare edge K-tile (crosses Klog)
                #pragma unroll
                for (int j = 0; j < 16; ++j) {
                    const int gk = kb + ak + j;
                    lA[ar * 40 + ak + j] = (gk < Klog) ? f2bf(arow[gk]) : (uint16_t)0;
                }
            }
        } else {
            const uint16_t* arow = Ah + aRowBase;
            const uint4 q0 = *(const uint4*)(arow + kb + ak);
            const uint4 q1 = *(const uint4*)(arow + kb + ak + 8);
            *(uint4*)(lA + ar * 40 + ak)     = q0;
            *(uint4*)(lA + ar * 40 + ak + 8) = q1;
        }
        // ---- stage B (already [N][K] in global) ----
        if constexpr (NTILE == 128) {
            const int br = tid >> 1, bk = (tid & 1) * 16;   // 128 rows x 32 k
            const uint16_t* brow = Bp + (long long)br * ldbK + kb + bk;
            const uint4 q0 = *(const uint4*)(brow);
            const uint4 q1 = *(const uint4*)(brow + 8);
            *(uint4*)(lBt + br * 40 + bk)     = q0;
            *(uint4*)(lBt + br * 40 + bk + 8) = q1;
        } else {
            const int br = tid >> 2, bk = (tid & 3) * 8;    // 64 rows x 32 k
            const uint4 qb = *(const uint4*)(Bp + (long long)br * ldbK + kb + bk);
            *(uint4*)(lBt + br * 40 + bk) = qb;
        }
        __syncthreads();

        // ---- build fragments (ISA 7.12.2 16-bit layouts) ----
        Frag32B fa[2], fb[NFR];
        const int fr = lane & 15, fh = lane >> 4;
        #pragma unroll
        for (int mi = 0; mi < 2; ++mi) {
            const uint16_t* p = lA + (wm * 32 + mi * 16 + fr) * 40 + fh * 8;
            fa[mi].q[0] = *(const uint4*)(p);        // K = h*8 .. +7
            fa[mi].q[1] = *(const uint4*)(p + 16);   // K = 16+h*8 .. +7
        }
        #pragma unroll
        for (int ni = 0; ni < NFR; ++ni) {
            const uint16_t* p = lBt + (wn * (NTILE / 2) + ni * 16 + fr) * 40 + fh * 8;
            fb[ni].q[0] = *(const uint4*)(p);        // K = h*8 .. +7
            fb[ni].q[1] = *(const uint4*)(p + 16);   // K = 16+h*8 .. +7
        }

        #pragma unroll
        for (int mi = 0; mi < 2; ++mi)
            #pragma unroll
            for (int ni = 0; ni < NFR; ++ni)
                acc[mi][ni] = __builtin_amdgcn_wmma_f32_16x16x32_bf16(
                    false, fa[mi].v, false, fb[ni].v,
                    (short)0, acc[mi][ni], false, false);
        __syncthreads();
    }

    // ---- write out (+bias) and accumulate BN stats ----
    const int cf = lane & 15, rh = lane >> 4;
    #pragma unroll
    for (int ni = 0; ni < NFR; ++ni) {
        const int colLocal = colTile + wn * (NTILE / 2) + ni * 16 + cf;
        const float bv = bias ? bias[(long long)z * biasBatch + colLocal] : 0.0f;
        float csum = 0.0f, csq = 0.0f;
        #pragma unroll
        for (int mi = 0; mi < 2; ++mi) {
            #pragma unroll
            for (int i = 0; i < 8; ++i) {
                const long long gRow = tileM + wm * 32 + mi * 16 + rh * 8 + i;
                const float v = acc[mi][ni][i] + bv;
                Cq[(long long)z * cBatch + gRow * ldc + colLocal] = v;
                csum += v; csq += v * v;
            }
        }
        if (ssum) {
            csum += __shfl_xor(csum, 16, 32);
            csq  += __shfl_xor(csq, 16, 32);
            if (lane < 16) {
                atomicAdd(&ssum[(long long)z * statBatch + colLocal], csum);
                atomicAdd(&ssumsq[(long long)z * statBatch + colLocal], csq);
            }
        }
    }
}

// ---------------------------------------------------------------------------
// Weight conversion kernels (produce bf16, B matrices pre-transposed to [N][K])
// ---------------------------------------------------------------------------
__global__ __launch_bounds__(256)
void conv_w0_kernel(const float* __restrict__ w0, uint16_t* __restrict__ out)
{
    // w0 f32 [C][O][S=500]  ->  out bf16 [C][O(=N)][Sp=512(=K)], zero padded
    const int idx = blockIdx.x * 256 + threadIdx.x;       // Cc*Oo*Sp = 524288
    const int c = idx >> 15, rem = idx & 32767;
    const int o = rem >> 9, sp = rem & 511;
    out[idx] = (sp < Ss) ? f2bf(w0[((long long)c * Oo + o) * Ss + sp]) : (uint16_t)0;
}

__global__ __launch_bounds__(256)
void transpose_conv_kernel(const float* __restrict__ in, uint16_t* __restrict__ out,
                           int K, int N, int total)
{
    // in f32 [Z][K][N]  ->  out bf16 [Z][N][K]
    const int idx = blockIdx.x * 256 + threadIdx.x;
    if (idx >= total) return;
    const int nk = N * K;
    const int z = idx / nk, rem = idx - z * nk;
    const int n = rem / K, k = rem - n * K;
    out[idx] = f2bf(in[(long long)z * nk + (long long)k * N + n]);
}

// ---------------------------------------------------------------------------
// BN finalize: scale = g*rsqrt(var+eps), shift = beta - mean*scale
// ---------------------------------------------------------------------------
__global__ __launch_bounds__(256)
void bn_finalize_kernel(const float* __restrict__ s, const float* __restrict__ q,
                        const float* __restrict__ g, const float* __restrict__ b,
                        float* __restrict__ sc, float* __restrict__ sh,
                        float count, int L)
{
    const int i = blockIdx.x * 256 + threadIdx.x;
    if (i < L) {
        const float m   = s[i] / count;
        const float var = q[i] / count - m * m;
        const float scl = g[i] * rsqrtf(var + EPS);
        sc[i] = scl;
        sh[i] = b[i] - m * scl;
    }
}

// ---------------------------------------------------------------------------
// h0 apply: bf16(swish(BN(h0_pre)))
// ---------------------------------------------------------------------------
__global__ __launch_bounds__(256)
void apply_h0_kernel(const float* __restrict__ pre, const float* __restrict__ sc,
                     const float* __restrict__ sh, uint16_t* __restrict__ h0bf)
{
    const int idx = blockIdx.x * 256 + threadIdx.x;   // Bsz*Ff
    const int f = idx & (Ff - 1);
    h0bf[idx] = f2bf(swishf(pre[idx] * sc[f] + sh[f]));
}

// ---------------------------------------------------------------------------
// Gate logits + softmax: one block per batch row; 8 waves cover 16 (t,e) dots
// ---------------------------------------------------------------------------
__global__ __launch_bounds__(256)
void gates_kernel(const uint16_t* __restrict__ h0bf, const float* __restrict__ gate_w,
                  const float* __restrict__ gate_b, float* __restrict__ gates)
{
    __shared__ float logits[Tt * Ee];
    const int b = blockIdx.x;
    const int tid = threadIdx.x, wave = tid >> 5, lane = tid & 31;
    const uint16_t* hrow = h0bf + (long long)b * Ff;

    for (int p = wave; p < Tt * Ee; p += 8) {
        const int t = p >> 3, e = p & 7;
        const float* gw = gate_w + ((long long)t * Ff) * Ee + e;   // stride E per f
        float s = 0.0f;
        for (int f = lane; f < Ff; f += 32)
            s += bf2f(hrow[f]) * gw[(long long)f * Ee];
        #pragma unroll
        for (int off = 16; off > 0; off >>= 1) s += __shfl_down(s, off, 32);
        if (lane == 0) logits[p] = s + gate_b[t * Ee + e];
    }
    __syncthreads();
    if (tid < Tt) {
        const int t = tid;
        float mx = logits[t * Ee];
        #pragma unroll
        for (int e = 1; e < Ee; ++e) mx = fmaxf(mx, logits[t * Ee + e]);
        float den = 0.0f, ex[Ee];
        #pragma unroll
        for (int e = 0; e < Ee; ++e) { ex[e] = __expf(logits[t * Ee + e] - mx); den += ex[e]; }
        const float inv = 1.0f / den;
        #pragma unroll
        for (int e = 0; e < Ee; ++e)
            gates[((long long)t * Bsz + b) * Ee + e] = ex[e] * inv;
    }
}

// ---------------------------------------------------------------------------
// Expert mix: mix[t][b][d] = sum_e gates[t][b][e] * swish(BN(eh_pre[b][e][d]))
// ---------------------------------------------------------------------------
__global__ __launch_bounds__(256)
void mix_kernel(const float* __restrict__ ehpre, const float* __restrict__ scE,
                const float* __restrict__ shE, const float* __restrict__ gates,
                uint16_t* __restrict__ mixbf)
{
    const int idx = blockIdx.x * 256 + threadIdx.x;    // Bsz*Dd
    const int b = idx >> 9, d = idx & (Dd - 1);
    float a0 = 0.0f, a1 = 0.0f;
    #pragma unroll
    for (int e = 0; e < Ee; ++e) {
        const float pre = ehpre[((long long)b * Ee + e) * Dd + d];
        const float sw = swishf(pre * scE[e * Dd + d] + shE[e * Dd + d]);
        a0 += gates[(long long)b * Ee + e] * sw;
        a1 += gates[(long long)(Bsz + b) * Ee + e] * sw;
    }
    mixbf[(long long)b * Dd + d] = f2bf(a0);
    mixbf[(long long)Bsz * Dd + (long long)b * Dd + d] = f2bf(a1);
}

// ---------------------------------------------------------------------------
// Task apply (in place): th = swish(BN(th_pre))
// ---------------------------------------------------------------------------
__global__ __launch_bounds__(256)
void apply_th_kernel(float* __restrict__ th, const float* __restrict__ scT,
                     const float* __restrict__ shT)
{
    const int idx = blockIdx.x * 256 + threadIdx.x;    // Tt*Bsz*Dd = 2^23
    const int t = idx >> 22;
    const int k = idx & (Dd - 1);
    th[idx] = swishf(th[idx] * scT[t * Dd + k] + shT[t * Dd + k]);
}

// ---------------------------------------------------------------------------
// Final heads: out[b][0:32] = th0[b]·fin_w_a + b_a ; out[b][32:34] = th1·fin_w_b + b_b
// One wave per (b, j) output.
// ---------------------------------------------------------------------------
__global__ __launch_bounds__(256)
void final_kernel(const float* __restrict__ th, const float* __restrict__ fwa,
                  const float* __restrict__ fba, const float* __restrict__ fwb,
                  const float* __restrict__ fbb, float* __restrict__ out)
{
    const int gw = blockIdx.x * 8 + (threadIdx.x >> 5);
    const int lane = threadIdx.x & 31;
    const int b = gw / 34, j = gw - b * 34;
    const float* trow; const float* w; int stride; float bias;
    if (j < 32) { trow = th + (long long)b * Dd;                       w = fwa + j;        stride = 32; bias = fba[j]; }
    else        { trow = th + (long long)Bsz * Dd + (long long)b * Dd; w = fwb + (j - 32); stride = 2;  bias = fbb[j - 32]; }
    float s = 0.0f;
    for (int k = lane; k < Dd; k += 32) s += trow[k] * w[(long long)k * stride];
    #pragma unroll
    for (int off = 16; off > 0; off >>= 1) s += __shfl_down(s, off, 32);
    if (lane == 0) out[(long long)b * 34 + j] = s + bias;
}

// ---------------------------------------------------------------------------
// Launch
// ---------------------------------------------------------------------------
extern "C" void kernel_launch(void* const* d_in, const int* in_sizes, int n_in,
                              void* d_out, int out_size, void* d_ws, size_t ws_size,
                              hipStream_t stream)
{
    (void)in_sizes; (void)n_in; (void)out_size; (void)ws_size;
    const float* x       = (const float*)d_in[0];
    const float* w0      = (const float*)d_in[1];
    const float* b0      = (const float*)d_in[2];
    const float* bn0_g   = (const float*)d_in[3];
    const float* bn0_b   = (const float*)d_in[4];
    const float* gate_w  = (const float*)d_in[5];
    const float* gate_b  = (const float*)d_in[6];
    const float* exp_w   = (const float*)d_in[7];
    const float* exp_bng = (const float*)d_in[8];
    const float* exp_bnb = (const float*)d_in[9];
    const float* task_w  = (const float*)d_in[10];
    const float* task_bng= (const float*)d_in[11];
    const float* task_bnb= (const float*)d_in[12];
    const float* fin_wa  = (const float*)d_in[13];
    const float* fin_ba  = (const float*)d_in[14];
    const float* fin_wb  = (const float*)d_in[15];
    const float* fin_bb  = (const float*)d_in[16];
    float* out = (float*)d_out;

    char* base = (char*)d_ws;
    size_t off = 0;
    auto alloc = [&](size_t bytes) -> void* {
        void* p = base + off;
        off = (off + bytes + 255) & ~(size_t)255;
        return p;
    };
    uint16_t* w0bf    = (uint16_t*)alloc((size_t)Cc * Oo * Sp * 2);        // [C][N=64][K=512]
    uint16_t* expwt   = (uint16_t*)alloc((size_t)Ee * Dd * Ff * 2);        // [E][N=512][K=1024]
    uint16_t* taskwt  = (uint16_t*)alloc((size_t)Tt * Dd * Dd * 2);        // [T][N=512][K=512]
    float*    h0pre   = (float*)   alloc((size_t)Bsz * Ff * 4);
    uint16_t* h0bf    = (uint16_t*)alloc((size_t)Bsz * Ff * 2);
    float*    ehpre   = (float*)   alloc((size_t)Bsz * Ee * Dd * 4);
    uint16_t* mixbf   = (uint16_t*)alloc((size_t)Tt * Bsz * Dd * 2);
    float*    thpre   = (float*)   alloc((size_t)Tt * Bsz * Dd * 4);
    float*    gates   = (float*)   alloc((size_t)Tt * Bsz * Ee * 4);
    float*    stats   = (float*)   alloc((size_t)12288 * 4);               // zeroed
    float*    scales  = (float*)   alloc((size_t)12288 * 4);
    float *s0 = stats,        *q0 = stats + 1024;
    float *sE = stats + 2048, *qE = stats + 6144;
    float *sT = stats + 10240,*qT = stats + 11264;
    float *sc0 = scales,        *sh0 = scales + 1024;
    float *scE = scales + 2048, *shE = scales + 6144;
    float *scT = scales + 10240,*shT = scales + 11264;

    hipMemsetAsync(stats, 0, 12288 * sizeof(float), stream);

    // Weight conversion / pre-transposition
    conv_w0_kernel<<<(Cc * Oo * Sp) / 256, 256, 0, stream>>>(w0, w0bf);
    transpose_conv_kernel<<<(Ee * Ff * Dd + 255) / 256, 256, 0, stream>>>(
        exp_w, expwt, Ff, Dd, Ee * Ff * Dd);
    transpose_conv_kernel<<<(Tt * Dd * Dd + 255) / 256, 256, 0, stream>>>(
        task_w, taskwt, Dd, Dd, Tt * Dd * Dd);

    // 1) SplitLinear: 16 chunk GEMMs [8192x500]x[500x64] (K padded to 512) + stats
    gemm_bf16_wmma<true, 64><<<dim3(Bsz / 128, 1, Cc), 256, 0, stream>>>(
        x, INF, (long long)Ss,
        w0bf, Sp, (long long)Oo * Sp,
        h0pre, Ff, (long long)Oo,
        b0, Oo, s0, q0, Oo, Sp, Ss);
    bn_finalize_kernel<<<4, 256, 0, stream>>>(s0, q0, bn0_g, bn0_b, sc0, sh0, (float)Bsz, Ff);
    apply_h0_kernel<<<(Bsz * Ff) / 256, 256, 0, stream>>>(h0pre, sc0, sh0, h0bf);

    // 2) Gates (softmax over experts, per task)
    gates_kernel<<<Bsz, 256, 0, stream>>>(h0bf, gate_w, gate_b, gates);

    // 3) Expert GEMMs: 8 x ([8192x1024]x[1024x512]) + stats, 128x128 tiles
    gemm_bf16_wmma<false, 128><<<dim3(Bsz / 128, Dd / 128, Ee), 256, 0, stream>>>(
        h0bf, Ff, 0LL,
        expwt, Ff, (long long)Dd * Ff,
        ehpre, Ee * Dd, (long long)Dd,
        nullptr, 0, sE, qE, Dd, Ff, Ff);
    bn_finalize_kernel<<<16, 256, 0, stream>>>(sE, qE, exp_bng, exp_bnb, scE, shE, (float)Bsz, Ee * Dd);

    // 4) Gate-weighted expert mix -> bf16
    mix_kernel<<<(Bsz * Dd) / 256, 256, 0, stream>>>(ehpre, scE, shE, gates, mixbf);

    // 5) Task GEMMs: 2 x ([8192x512]x[512x512]) + stats, 128x128 tiles
    gemm_bf16_wmma<false, 128><<<dim3(Bsz / 128, Dd / 128, Tt), 256, 0, stream>>>(
        mixbf, Dd, (long long)Bsz * Dd,
        taskwt, Dd, (long long)Dd * Dd,
        thpre, Dd, (long long)Bsz * Dd,
        nullptr, 0, sT, qT, Dd, Dd, Dd);
    bn_finalize_kernel<<<4, 256, 0, stream>>>(sT, qT, task_bng, task_bnb, scT, shT, (float)Bsz, Tt * Dd);
    apply_th_kernel<<<(Tt * Bsz * Dd) / 256, 256, 0, stream>>>(thpre, scT, shT);

    // 6) Final heads -> out [8192 x 34]
    final_kernel<<<(Bsz * 34) / 8, 256, 0, stream>>>(thpre, fin_wa, fin_ba, fin_wb, fin_bb, out);
}
